// MHA_33440615366682
// MI455X (gfx1250) — matrix operand order repaired and checked by reading, not compiled, not run on hardware
//
#include <hip/hip_runtime.h>

// ---------------------------------------------------------------------------
// MHA forward for MI455X (gfx1250): bf16 WMMA everywhere, f32 accumulate.
// ---------------------------------------------------------------------------

typedef __attribute__((ext_vector_type(16))) __bf16 v16bf;
typedef __attribute__((ext_vector_type(8)))  __bf16 v8bf;
typedef __attribute__((ext_vector_type(8)))  float  v8f;

#define SEQ     2048
#define EMBED   1024
#define NHEAD   16
#define HDIM    64
#define BATCH   4
#define NEGV    (-10000.0f)

// ---- WMMA fragment loaders (layouts per CDNA5 ISA 7.12.2) ------
// A (16x32 bf16): lane L<16 -> row L, K chunks {kh..kh+7, kh+16..kh+23} with
// kh = 8*(L>=16). Two 16B loads per lane.
__device__ __forceinline__ v16bf frag_a(const __bf16* p, int ld, int lane) {
    int r  = lane & 15;
    int kh = (lane >> 4) * 8;
    const __bf16* q = p + r * ld + kh;
    v8bf lo = *(const v8bf*)(q);
    v8bf hi = *(const v8bf*)(q + 16);
    return __builtin_shufflevector(lo, hi, 0,1,2,3,4,5,6,7,8,9,10,11,12,13,14,15);
}

// B (32x16 bf16): lane L<16 -> column L, K chunks {kh..kh+7, kh+16..kh+23} with
// kh = 8*(L>=16).
// Source matrix stored row-major as [N][K] (column of B == row of source).
__device__ __forceinline__ v16bf frag_b(const __bf16* p, int ld, int lane) {
    int c  = lane & 15;
    int ko = (lane >> 4) * 8;
    const __bf16* q = p + c * ld + ko;
    v8bf lo = *(const v8bf*)(q);
    v8bf hi = *(const v8bf*)(q + 16);
    return __builtin_shufflevector(lo, hi, 0,1,2,3,4,5,6,7,8,9,10,11,12,13,14,15);
}

__device__ __forceinline__ v8f wmma_bf16(v16bf a, v16bf b, v8f c) {
    return __builtin_amdgcn_wmma_f32_16x16x32_bf16(false, a, false, b, (short)0, c,
                                                   false, false);
}

// ---------------------------------------------------------------------------
// Phase 1: f32 -> bf16 conversion
// ---------------------------------------------------------------------------
__global__ void cvt_f32_bf16(const float* __restrict__ in, __bf16* __restrict__ out,
                             int n) {
    int i = blockIdx.x * blockDim.x + threadIdx.x;
    int st = gridDim.x * blockDim.x;
    for (; i < n; i += st) out[i] = (__bf16)in[i];
}

// ---------------------------------------------------------------------------
// Phase 2/4: GEMM  C[M,N] = A[M,K] * B[N,K]^T + bias
//   mode 0: QKV epilogue -> scatter bf16 into Q[b,h,s,d], K[b,h,s,d]*0.125,
//           Vt[b,h,d,s]
//   mode 1: plain f32 output (row-major MxN)
// Block = 256 threads (8 waves); wave tile 64x64; block tile 128x256.
// ---------------------------------------------------------------------------
__global__ void __launch_bounds__(256)
gemm_bf16_wmma(const __bf16* __restrict__ A, const __bf16* __restrict__ B,
               int K, int N, const float* __restrict__ bias,
               float* __restrict__ outf,
               __bf16* __restrict__ Qb, __bf16* __restrict__ Kb,
               __bf16* __restrict__ Vt, int mode) {
    const int lane = threadIdx.x & 31;
    const int wave = threadIdx.x >> 5;
    const int m0 = blockIdx.y * 128 + (wave >> 2) * 64;
    const int n0 = blockIdx.x * 256 + (wave & 3) * 64;

    v8f acc[4][4];
#pragma unroll
    for (int i = 0; i < 4; ++i)
#pragma unroll
        for (int j = 0; j < 4; ++j) acc[i][j] = (v8f)(0.0f);

    for (int k = 0; k < K; k += 32) {
        v16bf af[4], bf[4];
#pragma unroll
        for (int ti = 0; ti < 4; ++ti)
            af[ti] = frag_a(A + (size_t)(m0 + ti * 16) * K + k, K, lane);
#pragma unroll
        for (int tj = 0; tj < 4; ++tj)
            bf[tj] = frag_b(B + (size_t)(n0 + tj * 16) * K + k, K, lane);
#pragma unroll
        for (int ti = 0; ti < 4; ++ti)
#pragma unroll
            for (int tj = 0; tj < 4; ++tj)
                acc[ti][tj] = wmma_bf16(af[ti], bf[tj], acc[ti][tj]);
    }

    const int rhi = (lane >> 4) * 8;
    const int cl  = lane & 15;
#pragma unroll
    for (int ti = 0; ti < 4; ++ti) {
#pragma unroll
        for (int tj = 0; tj < 4; ++tj) {
            int colb = n0 + tj * 16 + cl;
            float bs = bias[colb];
#pragma unroll
            for (int r = 0; r < 8; ++r) {
                int row = m0 + ti * 16 + r + rhi;
                float v = acc[ti][tj][r] + bs;
                if (mode == 1) {
                    outf[(size_t)row * N + colb] = v;
                } else {
                    int part = colb >> 10;          // 0=q 1=k 2=v
                    int rem  = colb & 1023;
                    int h    = rem >> 6;
                    int d    = rem & 63;
                    int b    = row >> 11;           // /2048
                    int s    = row & 2047;
                    if (part == 0) {
                        Qb[(((size_t)(b * NHEAD + h) * SEQ) + s) * HDIM + d] = (__bf16)v;
                    } else if (part == 1) {
                        Kb[(((size_t)(b * NHEAD + h) * SEQ) + s) * HDIM + d] =
                            (__bf16)(v * 0.125f);   // fold 1/sqrt(64)
                    } else {
                        Vt[(((size_t)(b * NHEAD + h) * HDIM) + d) * SEQ + s] = (__bf16)v;
                    }
                }
            }
        }
    }
}

// ---------------------------------------------------------------------------
// Phase 3: flash attention. One wave = 32 query rows of one (b,h).
// Block = 256 threads = 8 waves; blockIdx.x = q-chunk (256 rows), .y = b*h.
// KV tiles of 64; online softmax in f32; P staged via LDS (stride 72 bf16
// = 144B rows -> 16B aligned, conflict-free) and reloaded as A-fragments.
// ---------------------------------------------------------------------------
#define LDS_STRIDE 72

__global__ void __launch_bounds__(256)
flash_attn_wmma(const __bf16* __restrict__ Qb, const __bf16* __restrict__ Kb,
                const __bf16* __restrict__ Vt, const unsigned char* __restrict__ kpm,
                __bf16* __restrict__ ctxb) {
    __shared__ __align__(16) __bf16 plds[8][32][LDS_STRIDE];

    const int lane = threadIdx.x & 31;
    const int wave = threadIdx.x >> 5;
    const int bh = blockIdx.y;
    const int b = bh >> 4, h = bh & 15;
    const int qbase = blockIdx.x * 256 + wave * 32;

    const __bf16* Qw = Qb + ((size_t)(b * NHEAD + h) * SEQ) * HDIM;
    const __bf16* Kw = Kb + ((size_t)(b * NHEAD + h) * SEQ) * HDIM;
    const __bf16* Vw = Vt + ((size_t)(b * NHEAD + h) * HDIM) * SEQ;

    const int rhi = (lane >> 4) * 8;
    const int cl  = lane & 15;

    v8f ctx[2][4];
#pragma unroll
    for (int i = 0; i < 2; ++i)
#pragma unroll
        for (int j = 0; j < 4; ++j) ctx[i][j] = (v8f)(0.0f);
    float m_i[2][8], l_i[2][8];
#pragma unroll
    for (int i = 0; i < 2; ++i)
#pragma unroll
        for (int r = 0; r < 8; ++r) { m_i[i][r] = -3.0e38f; l_i[i][r] = 0.0f; }

    // keep Q fragments resident (2 row-tiles x 2 K-steps)
    v16bf qf[2][2];
#pragma unroll
    for (int ti = 0; ti < 2; ++ti)
#pragma unroll
        for (int ks = 0; ks < 2; ++ks)
            qf[ti][ks] = frag_a(Qw + (size_t)(qbase + ti * 16) * HDIM + ks * 32,
                                HDIM, lane);

    for (int s0 = 0; s0 <= qbase + 31; s0 += 64) {
        // ---- S = Q * K^T (scale already folded into Kb) ----
        v8f sacc[2][4];
#pragma unroll
        for (int i = 0; i < 2; ++i)
#pragma unroll
            for (int j = 0; j < 4; ++j) sacc[i][j] = (v8f)(0.0f);
#pragma unroll
        for (int ks = 0; ks < 2; ++ks) {
            v16bf kf[4];
#pragma unroll
            for (int tj = 0; tj < 4; ++tj)
                kf[tj] = frag_b(Kw + (size_t)(s0 + tj * 16) * HDIM + ks * 32,
                                HDIM, lane);
#pragma unroll
            for (int ti = 0; ti < 2; ++ti)
#pragma unroll
                for (int tj = 0; tj < 4; ++tj)
                    sacc[ti][tj] = wmma_bf16(qf[ti][ks], kf[tj], sacc[ti][tj]);
        }

        // ---- padding mask for this lane's columns ----
        float pad[4];
#pragma unroll
        for (int tj = 0; tj < 4; ++tj) {
            int c = s0 + tj * 16 + cl;
            pad[tj] = kpm[b * SEQ + c] ? 0.0f : NEGV;
        }

        // ---- online softmax ----
#pragma unroll
        for (int ti = 0; ti < 2; ++ti) {
#pragma unroll
            for (int r = 0; r < 8; ++r) {
                int row = qbase + ti * 16 + r + rhi;
                float sv[4];
                float smax = -3.0e38f;
#pragma unroll
                for (int tj = 0; tj < 4; ++tj) {
                    int c = s0 + tj * 16 + cl;
                    float v = sacc[ti][tj][r] + pad[tj];
                    if (c > row) v += NEGV;       // causal (additive, as in ref)
                    sv[tj] = v;
                    smax = fmaxf(smax, v);
                }
#pragma unroll
                for (int off = 1; off < 16; off <<= 1)
                    smax = fmaxf(smax, __shfl_xor(smax, off, 32));
                float mnew  = fmaxf(m_i[ti][r], smax);
                float alpha = __expf(m_i[ti][r] - mnew);
                float rs = 0.0f;
#pragma unroll
                for (int tj = 0; tj < 4; ++tj) {
                    float p = __expf(sv[tj] - mnew);
                    rs += p;
                    plds[wave][ti * 16 + r + rhi][tj * 16 + cl] = (__bf16)p;
                }
#pragma unroll
                for (int off = 1; off < 16; off <<= 1)
                    rs += __shfl_xor(rs, off, 32);
                l_i[ti][r] = l_i[ti][r] * alpha + rs;
                m_i[ti][r] = mnew;
#pragma unroll
                for (int dj = 0; dj < 4; ++dj)
                    ctx[ti][dj][r] *= alpha;
            }
        }

        // ---- ctx += P * V  (A from LDS, B from Vt[b,h,d,s] contiguous) ----
#pragma unroll
        for (int ks = 0; ks < 2; ++ks) {
            v16bf pf[2], vf[4];
#pragma unroll
            for (int ti = 0; ti < 2; ++ti)
                pf[ti] = frag_a(&plds[wave][ti * 16][ks * 32], LDS_STRIDE, lane);
#pragma unroll
            for (int dj = 0; dj < 4; ++dj)
                vf[dj] = frag_b(Vw + (size_t)(dj * 16) * SEQ + s0 + ks * 32,
                                SEQ, lane);
#pragma unroll
            for (int ti = 0; ti < 2; ++ti)
#pragma unroll
                for (int dj = 0; dj < 4; ++dj)
                    ctx[ti][dj] = wmma_bf16(pf[ti], vf[dj], ctx[ti][dj]);
        }
    }

    // ---- normalize and store ctx as bf16 [b*s, h*64+d] ----
#pragma unroll
    for (int ti = 0; ti < 2; ++ti) {
#pragma unroll
        for (int r = 0; r < 8; ++r) {
            float inv = 1.0f / l_i[ti][r];
            size_t row = (size_t)b * SEQ + qbase + ti * 16 + r + rhi;
#pragma unroll
            for (int dj = 0; dj < 4; ++dj) {
                int col = h * HDIM + dj * 16 + cl;
                ctxb[row * EMBED + col] = (__bf16)(ctx[ti][dj][r] * inv);
            }
        }
    }
}

// ---------------------------------------------------------------------------
extern "C" void kernel_launch(void* const* d_in, const int* in_sizes, int n_in,
                              void* d_out, int out_size, void* d_ws, size_t ws_size,
                              hipStream_t stream) {
    const float*         x    = (const float*)d_in[0];
    const unsigned char* kpm  = (const unsigned char*)d_in[1];
    const float*         wqkv = (const float*)d_in[2];
    const float*         bqkv = (const float*)d_in[3];
    const float*         wout = (const float*)d_in[4];
    const float*         bout = (const float*)d_in[5];
    float*               out  = (float*)d_out;

    const int M = BATCH * SEQ;                 // 8192
    size_t off = 0;
    auto carve = [&](size_t elems) {
        __bf16* p = (__bf16*)((char*)d_ws + off);
        off += (elems * sizeof(__bf16) + 255) & ~(size_t)255;
        return p;
    };
    __bf16* xb    = carve((size_t)M * EMBED);
    __bf16* wqb   = carve((size_t)3 * EMBED * EMBED);
    __bf16* wob   = carve((size_t)EMBED * EMBED);
    __bf16* Qb    = carve((size_t)M * EMBED);
    __bf16* Kb    = carve((size_t)M * EMBED);
    __bf16* Vt    = carve((size_t)M * EMBED);
    __bf16* ctxb  = carve((size_t)M * EMBED);
    (void)ws_size; (void)n_in; (void)in_sizes; (void)out_size;

    // Phase 1: conversions
    cvt_f32_bf16<<<4096, 256, 0, stream>>>(x, xb, M * EMBED);
    cvt_f32_bf16<<<4096, 256, 0, stream>>>(wqkv, wqb, 3 * EMBED * EMBED);
    cvt_f32_bf16<<<2048, 256, 0, stream>>>(wout, wob, EMBED * EMBED);

    // Phase 2: QKV projection (M x 3072, K=1024), scatter into Q/K/Vt
    gemm_bf16_wmma<<<dim3(3 * EMBED / 256, M / 128), 256, 0, stream>>>(
        xb, wqb, EMBED, 3 * EMBED, bqkv, nullptr, Qb, Kb, Vt, 0);

    // Phase 3: flash attention
    flash_attn_wmma<<<dim3(SEQ / 256, BATCH * NHEAD), 256, 0, stream>>>(
        Qb, Kb, Vt, kpm, ctxb);

    // Phase 4: output projection (M x 1024, K=1024), f32 out + bias
    gemm_bf16_wmma<<<dim3(EMBED / 256, M / 128), 256, 0, stream>>>(
        ctxb, wob, EMBED, EMBED, bout, out, nullptr, nullptr, nullptr, 1);
}
